// OnTheFlyLinear_3461743641082
// MI455X (gfx1250) — matrix operand order repaired and checked by reading, not compiled, not run on hardware
//
#include <hip/hip_runtime.h>

// ---- CDNA5 (gfx1250) wave32 WMMA types ----
typedef __attribute__((ext_vector_type(16))) _Float16 v16h;
typedef __attribute__((ext_vector_type(4)))  _Float16 v4h;
typedef __attribute__((ext_vector_type(8)))  float    v8f;

#define TILE_M 128
#define TILE_N 128
#define TILE_K 32
#define K_DIM  4096
#define NKSTEPS (K_DIM / TILE_K)

#define A_STRIDE 36                      // dwords per raw-f32 A row (32 + 4 pad)
#define A_BUF_DW (TILE_M * A_STRIDE)     // dwords per A buffer
#define B_STRIDE 40                      // halves per f16 B row (32 + 8 pad)
#define B_BUF_H  (TILE_N * B_STRIDE)     // halves per B buffer

union FragU { v16h v; uint4 q[2]; };

__device__ __forceinline__ v4h pack4(float a, float b, float c, float d) {
    v4h r;
    r.x = (_Float16)a; r.y = (_Float16)b; r.z = (_Float16)c; r.w = (_Float16)d;
    return r;
}

__global__ __launch_bounds__(256)
void dequant_linear_wmma(const float* __restrict__ X,      // [M][K] fp32
                         const int*   __restrict__ W,      // [N][K] int8-valued int32
                         const float* __restrict__ scale,  // [1]
                         const float* __restrict__ bias,   // [N]
                         float*       __restrict__ Out,    // [M][N]
                         int M, int N)
{
    __shared__ float    AshRaw[2 * A_BUF_DW];   // raw fp32 x tiles (async-filled)
    __shared__ _Float16 Bsh[2 * B_BUF_H];       // f16 W tiles (register-staged)

    const int tid   = threadIdx.x;
    const int lane  = tid & 31;
    const int wave  = tid >> 5;      // 0..7
    const int waveM = wave & 3;      // 4 waves along M -> 32 rows each
    const int waveN = wave >> 2;     // 2 waves along N -> 64 cols each

    const int m0 = blockIdx.x * TILE_M;   // M fastest: blocks sharing W strip co-run
    const int n0 = blockIdx.y * TILE_N;

    // staging geometry: 256 threads x 4 chunks of 16B cover a 128x32-dword tile
    const int rowS = tid >> 3;            // 0..31 (+ r*32)
    const int c4   = (tid & 7) * 4;       // dword/half column 0,4,...,28

    // ---- fragment lane geometry (ISA 7.12.2, wave32) ----
    const int mrow   = lane & 15;
    const int khalfA = (lane >> 4) * 8;   // A: lanes 16-31 shift K by 8
    const int ncol   = lane & 15;
    const int kbB    = (lane >> 4) * 8;   // B: lanes 16-31 shift K by 8

    const unsigned long long xbase = (unsigned long long)X;
    const unsigned ldsA0 = (unsigned)(unsigned long long)(&AshRaw[0]);

    v8f acc[2][4];
#pragma unroll
    for (int mt = 0; mt < 2; ++mt)
#pragma unroll
        for (int nt = 0; nt < 4; ++nt)
            acc[mt][nt] = (v8f){0.f, 0.f, 0.f, 0.f, 0.f, 0.f, 0.f, 0.f};

    // ---- async copy of one raw x tile into LDS buffer b (GVS mode) ----
    auto issueA = [&](int b, int kk) {
#pragma unroll
        for (int r = 0; r < 4; ++r) {
            const int row  = rowS + r * 32;
            const unsigned lds  = ldsA0 + (unsigned)((b * A_BUF_DW + row * A_STRIDE + c4) * 4);
            const unsigned voff = (unsigned)(((m0 + row) * K_DIM + kk + c4) * 4);
            asm volatile("global_load_async_to_lds_b128 %0, %1, %2 offset:0"
                         :: "v"(lds), "v"(voff), "s"(xbase) : "memory");
        }
    };

    int4 wreg[4];
    auto loadB = [&](int kk) {
#pragma unroll
        for (int r = 0; r < 4; ++r)
            wreg[r] = *(const int4*)(W + (size_t)(n0 + rowS + r * 32) * K_DIM + kk + c4);
    };
    auto storeB = [&](int b) {
#pragma unroll
        for (int r = 0; r < 4; ++r) {
            const int row = rowS + r * 32;
            *(v4h*)&Bsh[b * B_BUF_H + row * B_STRIDE + c4] =
                pack4((float)wreg[r].x, (float)wreg[r].y,
                      (float)wreg[r].z, (float)wreg[r].w);
        }
    };

    // ---- prologue: stage k-step 0 ----
    issueA(0, 0);
    loadB(0);

    for (int j = 0; j < NKSTEPS; ++j) {
        const int buf = j & 1;

        // 1. convert staged W regs -> f16 LDS (current buffer)
        storeB(buf);

        // 2. drain our async x-tile writes, then block-wide fence
        asm volatile("s_wait_asynccnt 0x0" ::: "memory");
        __syncthreads();

        // 3. kick off next k-step's async x tile (buffer just freed by barrier)
        if (j + 1 < NKSTEPS)
            issueA(buf ^ 1, (j + 1) * TILE_K);

        // 4a. A fragments: raw f32 from LDS -> f16 in-register
        const float* Ab = &AshRaw[buf * A_BUF_DW];
        v16h af[2];
#pragma unroll
        for (int mt = 0; mt < 2; ++mt) {
            const int row = waveM * 32 + mt * 16 + mrow;
            const float4 f0 = *(const float4*)&Ab[row * A_STRIDE + khalfA];
            const float4 f1 = *(const float4*)&Ab[row * A_STRIDE + khalfA + 4];
            const float4 f2 = *(const float4*)&Ab[row * A_STRIDE + 16 + khalfA];
            const float4 f3 = *(const float4*)&Ab[row * A_STRIDE + 16 + khalfA + 4];
            v16h a;
            a[0]  = (_Float16)f0.x; a[1]  = (_Float16)f0.y;
            a[2]  = (_Float16)f0.z; a[3]  = (_Float16)f0.w;
            a[4]  = (_Float16)f1.x; a[5]  = (_Float16)f1.y;
            a[6]  = (_Float16)f1.z; a[7]  = (_Float16)f1.w;
            a[8]  = (_Float16)f2.x; a[9]  = (_Float16)f2.y;
            a[10] = (_Float16)f2.z; a[11] = (_Float16)f2.w;
            a[12] = (_Float16)f3.x; a[13] = (_Float16)f3.y;
            a[14] = (_Float16)f3.z; a[15] = (_Float16)f3.w;
            af[mt] = a;
        }

        // 4b. B fragments: 2x ds_load_b128 each (8 f16 at k=8h.., 8 at k=16+8h..)
        const _Float16* Bb = &Bsh[buf * B_BUF_H];
        FragU bf[4];
#pragma unroll
        for (int nt = 0; nt < 4; ++nt) {
            const int row = waveN * 64 + nt * 16 + ncol;
            bf[nt].q[0] = *(const uint4*)&Bb[row * B_STRIDE + kbB];
            bf[nt].q[1] = *(const uint4*)&Bb[row * B_STRIDE + kbB + 16];
        }

        // 5. prefetch next k-step's W ints (overlaps with WMMA below)
        if (j + 1 < NKSTEPS)
            loadB((j + 1) * TILE_K);

        // 6. 8x v_wmma_f32_16x16x32_f16
#pragma unroll
        for (int mt = 0; mt < 2; ++mt)
#pragma unroll
            for (int nt = 0; nt < 4; ++nt)
                acc[mt][nt] = __builtin_amdgcn_wmma_f32_16x16x32_f16(
                    false, af[mt], false, bf[nt].v,
                    (short)0, acc[mt][nt], false, false);
    }

    // ---- epilogue: out = acc*scale + bias ----
    const float s = scale[0];
#pragma unroll
    for (int mt = 0; mt < 2; ++mt) {
#pragma unroll
        for (int nt = 0; nt < 4; ++nt) {
            const int n = n0 + waveN * 64 + nt * 16 + ncol;
            const float bv = bias[n];
            const int mbase = m0 + waveM * 32 + mt * 16 + (lane >> 4) * 8;
#pragma unroll
            for (int v = 0; v < 8; ++v) {
                Out[(size_t)(mbase + v) * N + n] = acc[mt][nt][v] * s + bv;
            }
        }
    }
}

extern "C" void kernel_launch(void* const* d_in, const int* in_sizes, int n_in,
                              void* d_out, int out_size, void* d_ws, size_t ws_size,
                              hipStream_t stream) {
    const float* X     = (const float*)d_in[0];   // [2,512,4096] fp32
    const int*   W     = (const int*)d_in[1];     // [11008,4096] int32 (int8-valued)
    const float* scale = (const float*)d_in[2];   // [1]
    const float* bias  = (const float*)d_in[3];   // [11008]
    float*       Out   = (float*)d_out;           // [2,512,11008] fp32

    const int N = in_sizes[3];          // 11008
    const int M = in_sizes[0] / K_DIM;  // 1024

    dim3 grid(M / TILE_M, N / TILE_N);  // (8, 86): M-fastest -> W strip L2 reuse
    dim3 block(256);
    dequant_linear_wmma<<<grid, block, 0, stream>>>(X, W, scale, bias, Out, M, N);
}
